// ReferenceAttention_69260642615358
// MI455X (gfx1250) — matrix operand order repaired and checked by reading, not compiled, not run on hardware
//
#include <hip/hip_runtime.h>

#define BB 2
#define TT 8
#define HH 28
#define WWI 28
#define CC 768
#define NHEADS 12
#define CHD 64
#define NN (1 + HH * WWI)      // 785
#define HP 14
#define NP (1 + HP * HP)       // 197
#define LQ (TT * NP)           // 1576
#define MROWS (BB * LQ)        // 3152

typedef unsigned short u16;
typedef __attribute__((ext_vector_type(16))) __bf16 bf16x16;
typedef __attribute__((ext_vector_type(8)))  float  f32x8;

union FragA {
    bf16x16 v;
    uint4   q[2];
    u16     s[16];
};

__device__ __forceinline__ float bf2f(u16 h) {
    union { unsigned int u; float f; } c;
    c.u = ((unsigned int)h) << 16;
    return c.f;
}
__device__ __forceinline__ u16 f2bf(float f) {
    union { float f; unsigned int u; } c;
    c.f = f;
    unsigned int u = c.u;
    u += 0x7FFFu + ((u >> 16) & 1u);   // round to nearest even
    return (u16)(u >> 16);
}
__device__ __forceinline__ f32x8 wmma_bf16(const FragA& a, const FragA& b, f32x8 c) {
    return __builtin_amdgcn_wmma_f32_16x16x32_bf16(false, a.v, false, b.v,
                                                   (short)0, c, false, false);
}

// ---------------------------------------------------------------------------
// float -> bf16 conversion (weights)
// ---------------------------------------------------------------------------
__global__ void f2bf_kernel(const float* __restrict__ src, u16* __restrict__ dst, int n) {
    int i = blockIdx.x * 256 + threadIdx.x;
    if (i < n) dst[i] = f2bf(src[i]);
}

// ---------------------------------------------------------------------------
// Depthwise 3x3 stride-2 pool (body tokens) + per-head LayerNorm over 64 ch.
// One 64-lane group = one (bt, np, nh) unit. Output bf16, (B*T, NP, C).
// ---------------------------------------------------------------------------
__global__ void pool_norm_kernel(const float* __restrict__ x, const float* __restrict__ w,
                                 const float* __restrict__ g, const float* __restrict__ bta,
                                 u16* __restrict__ dst) {
    __shared__ float s_sum[256];
    __shared__ float s_sq[256];
    int tid = threadIdx.x;
    int ch  = tid & 63;
    int grp = tid >> 6;
    long long unit = (long long)blockIdx.x * 4 + grp;   // ordered (bt, np, nh)
    int nh = (int)(unit % NHEADS);
    long long u2 = unit / NHEADS;
    int np = (int)(u2 % NP);
    int bt = (int)(u2 / NP);

    float val = 0.f;
    if (np == 0) {
        val = x[((long long)bt * NN) * CC + nh * CHD + ch];
    } else {
        int i = (np - 1) / HP, j = (np - 1) % HP;
        for (int kh = 0; kh < 3; ++kh) {
            int ih = 2 * i - 1 + kh;
            if (ih < 0 || ih >= HH) continue;
            for (int kw = 0; kw < 3; ++kw) {
                int iw = 2 * j - 1 + kw;
                if (iw < 0 || iw >= WWI) continue;
                int n = 1 + ih * WWI + iw;
                val += x[((long long)bt * NN + n) * CC + nh * CHD + ch] *
                       w[(kh * 3 + kw) * CHD + ch];
            }
        }
    }
    s_sum[tid] = val;
    s_sq[tid]  = val * val;
    __syncthreads();
    for (int s = 32; s >= 1; s >>= 1) {
        if (ch < s) { s_sum[tid] += s_sum[tid + s]; s_sq[tid] += s_sq[tid + s]; }
        __syncthreads();
    }
    float mu   = s_sum[tid - ch] * (1.f / 64.f);
    float var  = s_sq[tid - ch] * (1.f / 64.f) - mu * mu;
    float rstd = rsqrtf(var + 1e-5f);
    float outv = (val - mu) * rstd * g[ch] + bta[ch];
    dst[((long long)bt * NP + np) * CC + nh * CHD + ch] = f2bf(outv);
}

// ---------------------------------------------------------------------------
// GEMM: D = A(M x 768) @ W^T, W is (768 x 768) row-major bf16 (rows = out cols).
// Each wave computes a 16(M) x 64(N) tile via v_wmma_f32_16x16x32_bf16.
// mode 0: scatter bf16 into (B, NH, LQ, 64); mode 1: f32 + bias into d_out.
// ---------------------------------------------------------------------------
__global__ void gemm_kernel(const u16* __restrict__ A, const u16* __restrict__ Wb,
                            int M, int njobs, int mode,
                            u16* __restrict__ outS, float* __restrict__ outF,
                            const float* __restrict__ bias) {
    int lane = threadIdx.x & 31;
    int wv   = threadIdx.x >> 5;
    int job  = blockIdx.x * 8 + wv;
    if (job >= njobs) return;
    int tm  = job / 12;
    int tn4 = job % 12;

    f32x8 acc[4];
    for (int n = 0; n < 4; ++n)
        for (int i = 0; i < 8; ++i) acc[n][i] = 0.f;

    int row  = tm * 16 + (lane & 15);
    int rowc = row < M ? row : M - 1;
    const u16* Arow = A + (long long)rowc * CC;
    int kb  = (lane >> 4) * 8;    // A-frag K base
    int kb2 = (lane >> 4) * 8;    // B-frag K base

    for (int k0 = 0; k0 < CC; k0 += 32) {
        FragA af;
        af.q[0] = *(const uint4*)(Arow + k0 + kb);
        af.q[1] = *(const uint4*)(Arow + k0 + kb + 16);
        for (int n = 0; n < 4; ++n) {
            int ncol = tn4 * 64 + n * 16 + (lane & 15);
            const u16* Wrow = Wb + (long long)ncol * CC + k0 + kb2;
            FragA bf;
            bf.q[0] = *(const uint4*)(Wrow);
            bf.q[1] = *(const uint4*)(Wrow + 16);
            acc[n] = wmma_bf16(af, bf, acc[n]);
        }
    }

    int mlo = tm * 16 + 8 * (lane >> 4);
    if (mode == 0) {
        for (int n = 0; n < 4; ++n) {
            int col = tn4 * 64 + n * 16 + (lane & 15);
            int nh = col >> 6, chn = col & 63;
            for (int v = 0; v < 8; ++v) {
                int m = mlo + v;
                if (m < M) {
                    int b = m / LQ, r = m % LQ;
                    outS[(((long long)b * NHEADS + nh) * LQ + r) * CHD + chn] =
                        f2bf(acc[n][v]);
                }
            }
        }
    } else {
        for (int n = 0; n < 4; ++n) {
            int col  = tn4 * 64 + n * 16 + (lane & 15);
            float bc = bias[col];
            for (int v = 0; v < 8; ++v) {
                int m = mlo + v;
                if (m < M) outF[(long long)m * CC + col] = acc[n][v] + bc;
            }
        }
    }
}

// ---------------------------------------------------------------------------
// Flash attention, one (b, h) per blockIdx.{z,y}, 64 query rows per block
// (4 waves x 16 rows). K/V streamed in 32-key chunks through LDS; online
// softmax; P converted D-layout -> A-layout via per-wave LDS round trip.
// ---------------------------------------------------------------------------
__global__ void attn_kernel(const u16* __restrict__ Qp, const u16* __restrict__ Kp,
                            const u16* __restrict__ Vp, u16* __restrict__ AO) {
    __shared__ __align__(16) u16 sK[32 * 64];       // K chunk, key-major
    __shared__ __align__(16) u16 sVt[64 * 32];      // V chunk transposed, d-major
    __shared__ __align__(16) u16 sP[4 * 16 * 32];   // per-wave P buffers

    int tid  = threadIdx.x;
    int lane = tid & 31;
    int wv   = tid >> 5;
    int b = blockIdx.z, h = blockIdx.y;
    const u16* Qh = Qp + ((long long)b * NHEADS + h) * LQ * CHD;
    const u16* Kh = Kp + ((long long)b * NHEADS + h) * LQ * CHD;
    const u16* Vh = Vp + ((long long)b * NHEADS + h) * LQ * CHD;
    int qbase = blockIdx.x * 64 + wv * 16;

    // resident Q fragments (two K-chunks of the 64-wide head dim)
    int rA = qbase + (lane & 15); if (rA >= LQ) rA = LQ - 1;
    int kb = (lane >> 4) * 8;
    FragA qf0, qf1;
    qf0.q[0] = *(const uint4*)(Qh + (long long)rA * CHD + kb);
    qf0.q[1] = *(const uint4*)(Qh + (long long)rA * CHD + kb + 16);
    qf1.q[0] = *(const uint4*)(Qh + (long long)rA * CHD + 32 + kb);
    qf1.q[1] = *(const uint4*)(Qh + (long long)rA * CHD + 32 + kb + 16);

    f32x8 Ot[4];
    for (int j = 0; j < 4; ++j)
        for (int i = 0; i < 8; ++i) Ot[j][i] = 0.f;
    float m_r[8], l_r[8];
    for (int v = 0; v < 8; ++v) { m_r[v] = -3.0e38f; l_r[v] = 0.f; }
    u16* myP = sP + wv * 512;
    int kb2 = (lane >> 4) * 8;

    for (int kbase = 0; kbase < LQ; kbase += 32) {
        __syncthreads();   // previous iteration done with sK/sVt
        // cooperative stage of K (row-major) and V^T (d-major)
        for (int part = 0; part < 2; ++part) {
            int e = (part * 128 + tid) * 8;
            int key = e >> 6, d0 = e & 63;
            uint4 kv, vvv;
            if (kbase + key < LQ) {
                kv  = *(const uint4*)(Kh + (long long)(kbase + key) * CHD + d0);
                vvv = *(const uint4*)(Vh + (long long)(kbase + key) * CHD + d0);
            } else {
                kv = make_uint4(0, 0, 0, 0); vvv = kv;
            }
            *(uint4*)(sK + e) = kv;
            union { uint4 q; u16 s[8]; } cv; cv.q = vvv;
            for (int i = 0; i < 8; ++i) sVt[(d0 + i) * 32 + key] = cv.s[i];
        }
        __syncthreads();

        // S = Q @ K^T : two 16-key accumulator tiles
        f32x8 st[2];
        for (int kt = 0; kt < 2; ++kt) {
            int keyr = kt * 16 + (lane & 15);
            f32x8 a; for (int i = 0; i < 8; ++i) a[i] = 0.f;
            FragA kf;
            kf.q[0] = *(const uint4*)(sK + keyr * 64 + kb2);
            kf.q[1] = *(const uint4*)(sK + keyr * 64 + kb2 + 16);
            a = wmma_bf16(qf0, kf, a);
            kf.q[0] = *(const uint4*)(sK + keyr * 64 + 32 + kb2);
            kf.q[1] = *(const uint4*)(sK + keyr * 64 + 32 + kb2 + 16);
            a = wmma_bf16(qf1, kf, a);
            st[kt] = a;
        }

        // scale + mask + online softmax (rows live across 16-lane halves)
        for (int v = 0; v < 8; ++v) {
            float x0 = st[0][v] * 0.125f, x1 = st[1][v] * 0.125f;
            int c0 = kbase + (lane & 15);
            if (c0 >= LQ)      x0 = -1.0e30f;
            if (c0 + 16 >= LQ) x1 = -1.0e30f;
            float mx = fmaxf(x0, x1);
            for (int off = 1; off < 16; off <<= 1)
                mx = fmaxf(mx, __shfl_xor(mx, off, 32));
            float mnew  = fmaxf(m_r[v], mx);
            float alpha = __expf(m_r[v] - mnew);
            float p0 = __expf(x0 - mnew);
            float p1 = __expf(x1 - mnew);
            st[0][v] = p0; st[1][v] = p1;
            float rs = p0 + p1;
            for (int off = 1; off < 16; off <<= 1)
                rs += __shfl_xor(rs, off, 32);
            l_r[v] = l_r[v] * alpha + rs;
            m_r[v] = mnew;
            for (int j = 0; j < 4; ++j) Ot[j][v] *= alpha;
        }

        // P: D-layout -> LDS -> A-layout (same-wave DS ops are in-order)
        int prow = 8 * (lane >> 4);
        for (int kt = 0; kt < 2; ++kt)
            for (int v = 0; v < 8; ++v)
                myP[(prow + v) * 32 + kt * 16 + (lane & 15)] = f2bf(st[kt][v]);
        FragA pf;
        int pr = lane & 15;
        pf.q[0] = *(const uint4*)(myP + pr * 32 + kb);
        pf.q[1] = *(const uint4*)(myP + pr * 32 + kb + 16);

        // O += P @ V  (B-frag rows are contiguous in sVt)
        for (int j = 0; j < 4; ++j) {
            int dcol = j * 16 + (lane & 15);
            FragA vf;
            vf.q[0] = *(const uint4*)(sVt + dcol * 32 + kb2);
            vf.q[1] = *(const uint4*)(sVt + dcol * 32 + kb2 + 16);
            Ot[j] = wmma_bf16(pf, vf, Ot[j]);
        }
    }

    // epilogue: normalize, residual q for global row >= 1, store bf16 (B, L, C)
    for (int j = 0; j < 4; ++j) {
        int d = j * 16 + (lane & 15);
        for (int v = 0; v < 8; ++v) {
            int r = qbase + 8 * (lane >> 4) + v;
            if (r < LQ) {
                float o = Ot[j][v] / l_r[v];
                if (r >= 1) o += bf2f(Qh[(long long)r * CHD + d]);
                AO[((long long)b * LQ + r) * CC + h * CHD + d] = f2bf(o);
            }
        }
    }
}

// ---------------------------------------------------------------------------
extern "C" void kernel_launch(void* const* d_in, const int* in_sizes, int n_in,
                              void* d_out, int out_size, void* d_ws, size_t ws_size,
                              hipStream_t stream) {
    (void)in_sizes; (void)n_in; (void)out_size; (void)ws_size;
    const float* x     = (const float*)d_in[0];
    const float* x_ref = (const float*)d_in[1];
    const float* wq    = (const float*)d_in[2];
    const float* wk    = (const float*)d_in[3];
    const float* wv    = (const float*)d_in[4];
    const float* wproj = (const float*)d_in[5];
    const float* bproj = (const float*)d_in[6];
    const float* cq    = (const float*)d_in[7];
    const float* ck    = (const float*)d_in[8];
    const float* cv    = (const float*)d_in[9];
    const float* gq    = (const float*)d_in[10];
    const float* bq    = (const float*)d_in[11];
    const float* gk    = (const float*)d_in[12];
    const float* bk    = (const float*)d_in[13];
    const float* gv    = (const float*)d_in[14];
    const float* bv    = (const float*)d_in[15];

    char* ws = (char*)d_ws;
    const size_t WSZ  = (size_t)CC * CC * 2;        // 1,179,648 B per bf16 weight
    const size_t TSZ  = (size_t)MROWS * CC * 2;     // 4,841,472 B per bf16 tensor
    u16* wqb = (u16*)(ws + 0 * WSZ);
    u16* wkb = (u16*)(ws + 1 * WSZ);
    u16* wvb = (u16*)(ws + 2 * WSZ);
    u16* wpb = (u16*)(ws + 3 * WSZ);
    char* p  = ws + 4 * WSZ;
    u16* qpn = (u16*)(p + 0 * TSZ);
    u16* kpn = (u16*)(p + 1 * TSZ);
    u16* vpn = (u16*)(p + 2 * TSZ);
    u16* Qpw = (u16*)(p + 3 * TSZ);
    u16* Kpw = (u16*)(p + 4 * TSZ);
    u16* Vpw = (u16*)(p + 5 * TSZ);
    u16* AO  = (u16*)(p + 6 * TSZ);

    const int WN = CC * CC;                 // 589824
    f2bf_kernel<<<(WN + 255) / 256, 256, 0, stream>>>(wq,    wqb, WN);
    f2bf_kernel<<<(WN + 255) / 256, 256, 0, stream>>>(wk,    wkb, WN);
    f2bf_kernel<<<(WN + 255) / 256, 256, 0, stream>>>(wv,    wvb, WN);
    f2bf_kernel<<<(WN + 255) / 256, 256, 0, stream>>>(wproj, wpb, WN);

    const int nUnits = (BB * TT) * NP * NHEADS;     // 37824, /4 per block
    pool_norm_kernel<<<nUnits / 4, 256, 0, stream>>>(x,     cq, gq, bq, qpn);
    pool_norm_kernel<<<nUnits / 4, 256, 0, stream>>>(x_ref, ck, gk, bk, kpn);
    pool_norm_kernel<<<nUnits / 4, 256, 0, stream>>>(x_ref, cv, gv, bv, vpn);

    const int njobs = ((MROWS + 15) / 16) * 12;     // 197 * 12 = 2364
    const int gblk  = (njobs + 7) / 8;              // 296 blocks of 8 waves
    gemm_kernel<<<gblk, 256, 0, stream>>>(qpn, wqb, MROWS, njobs, 0, Qpw, nullptr, nullptr);
    gemm_kernel<<<gblk, 256, 0, stream>>>(kpn, wkb, MROWS, njobs, 0, Kpw, nullptr, nullptr);
    gemm_kernel<<<gblk, 256, 0, stream>>>(vpn, wvb, MROWS, njobs, 0, Vpw, nullptr, nullptr);

    dim3 agrid((LQ + 63) / 64, NHEADS, BB);         // 25 x 12 x 2
    attn_kernel<<<agrid, 128, 0, stream>>>(Qpw, Kpw, Vpw, AO);

    gemm_kernel<<<gblk, 256, 0, stream>>>(AO, wpb, MROWS, njobs, 1,
                                          nullptr, (float*)d_out, bproj);
}
